// MultiStageVectorQuantization_38895223833113
// MI455X (gfx1250) — matrix operand order repaired and checked by reading, not compiled, not run on hardware
//
#include <hip/hip_runtime.h>
#include <hip/hip_bf16.h>

// Problem constants (match reference)
#define N_TOK  131072
#define DIM    64
#define QSTAGE 8
#define KCB    1024

typedef __attribute__((ext_vector_type(16))) __bf16         v16bf;
typedef __attribute__((ext_vector_type(8)))  float          v8f;
typedef __attribute__((ext_vector_type(8)))  unsigned short u16x8;
typedef __attribute__((ext_vector_type(16))) unsigned short v16u;
typedef __attribute__((ext_vector_type(4)))  float          f4;
typedef __attribute__((ext_vector_type(4)))  unsigned int   u32x4;
typedef __attribute__((ext_vector_type(4)))  int            i32x4;
typedef __attribute__((ext_vector_type(8)))  int            i32x8;

union A16 { v16bf bf; u16x8 h[2]; };   // 16B-aligned halves of an A fragment
union C8  { v8f  f;  f4    h[2]; };    // 16B-aligned halves of a C fragment
union CU  { v8f  f;  unsigned u[8]; }; // reinterpret accumulator as u32 keys

// LDS layout: two ping-pong half-stage buffers (256 KB total):
//   per buffer: A: 32 tiles * 32 lanes * 64B data, lane stride padded ->80B
//               C: 32 tiles * 32 lanes * 32B data, lane stride padded ->48B
#define A_LANE_STRIDE 80
#define A_TILE_STRIDE 2560          // 32 * 80
#define C_LANE_STRIDE 48
#define C_TILE_STRIDE 1536          // 32 * 48
#define AH_BYTES      81920         // 32 tiles * 2560
#define CH_BYTES      49152         // 32 tiles * 1536
#define BUF_BYTES     (AH_BYTES + CH_BYTES)   // 131072
#define SMEM_BYTES    (2 * BUF_BYTES)         // 262144

__device__ __forceinline__ unsigned short f2bf(float f) {
  // round-to-nearest-even f32 -> bf16 (finite inputs)
  unsigned u = __builtin_bit_cast(unsigned, f);
  unsigned r = u + 0x7FFFu + ((u >> 16) & 1u);
  return (unsigned short)(r >> 16);
}

__device__ __forceinline__ unsigned umin32(unsigned a, unsigned b) {
  return a < b ? a : b;
}

// ---------------------------------------------------------------------------
// TDM: 1-D contiguous global -> LDS with periodic padding (D# per ISA §8.3/8.4)
//   ndw elements of 4B; LDS gets padAmount(+1) DWORDs inserted after every
//   2^(padIntervalSel+1) DWORDs stored.
// ---------------------------------------------------------------------------
__device__ __forceinline__ void tdm_load_pad_1d(
    unsigned ldsAddr, const void* gptr, unsigned ndw,
    unsigned padIntervalSel, unsigned padAmountSel) {
  unsigned long long ga = (unsigned long long)(size_t)gptr;
  u32x4 g0;
  g0[0] = 1u;                                    // count=1, is_restore=0
  g0[1] = ldsAddr;                               // lds_addr (bytes)
  g0[2] = (unsigned)ga;                          // global_addr[31:0]
  g0[3] = (unsigned)((ga >> 32) & 0x01FFFFFFu)   // global_addr[56:32]
          | (2u << 30);                          // type = 2 ("image")
  i32x8 g1;
  g1[0] = (int)((2u << 16)                       // data_size = 4B
                | (1u << 20)                     // pad_enable
                | (padIntervalSel << 22)         // pad_interval
                | (padAmountSel << 25));         // pad_amount
  g1[1] = (int)((ndw & 0xFFFFu) << 16);          // tensor_dim0[15:0] @63:48
  g1[2] = (int)(((ndw >> 16) & 0xFFFFu)          // tensor_dim0[31:16]
                | (1u << 16));                   // tensor_dim1 = 1
  g1[3] = (int)((ndw & 0xFFFFu) << 16);          // tile_dim0 @127:112
  g1[4] = 0;                                     // tile_dim1=0, tile_dim2=0
  g1[5] = (int)ndw;                              // tensor_dim0_stride[31:0]
  g1[6] = 0;
  g1[7] = 0;
  i32x4 gz = (i32x4){0, 0, 0, 0};
#if defined(__clang_major__) && (__clang_major__ >= 23)
  i32x8 gz8 = (i32x8){0, 0, 0, 0, 0, 0, 0, 0};
  __builtin_amdgcn_tensor_load_to_lds(g0, g1, gz, gz, gz8, 0);
#else
  __builtin_amdgcn_tensor_load_to_lds(g0, g1, gz, gz, 0);
#endif
}

// Issue both TDM loads (A + C fragments) for half-stage h into buffer `buf`.
__device__ __forceinline__ void tdm_issue_half(
    int h, char* buf, const unsigned short* afrag, const float* cnf) {
  const int q = h >> 1, hf = h & 1;
  unsigned aLds = (unsigned)(size_t)buf;          // low 32 bits = LDS offset
  unsigned cLds = (unsigned)(size_t)(buf + AH_BYTES);
  const void* aG = (const void*)(afrag + (size_t)q * 65536 + (size_t)hf * 32768);
  const void* cG = (const void*)(cnf   + (size_t)q * 16384 + (size_t)hf * 8192);
  tdm_load_pad_1d(aLds, aG, 16384, 3u, 3u);  // 64KB, +4DW pad every 16DW (64B)
  tdm_load_pad_1d(cLds, cG,  8192, 2u, 3u);  // 32KB, +4DW pad every  8DW (32B)
}

// ---------------------------------------------------------------------------
// Prep 1: swizzle codebooks (f32 [Q][K][D]) into WMMA A-fragment order as
// bf16 with the -2 factor baked in (distance = ||r||^2 + ||c||^2 - 2 r.c).
// ---------------------------------------------------------------------------
__global__ __launch_bounds__(256) void vq_prep_afrag(
    const float* __restrict__ cb, unsigned short* __restrict__ afrag) {
  int fid = blockIdx.x * 256 + threadIdx.x;            // 0 .. 524287
  int e = fid & 15;
  int j = (fid >> 4) & 1;
  int l = (fid >> 5) & 31;
  int t = (fid >> 10) & 63;
  int q = fid >> 16;
  int hi = l >> 4;
  int code = t * 16 + (l & 15);
  int d = j * 32 + hi * 16 + e;
  float v = -2.0f * cb[((size_t)(q * KCB + code)) * DIM + d];
  afrag[fid] = f2bf(v);
}

// ---------------------------------------------------------------------------
// Prep 2: per-code squared norms (+1.0 positivity bias), broadcast into WMMA
// C-fragment order. Also zeroes the 8 loss outputs.
// ---------------------------------------------------------------------------
__global__ __launch_bounds__(256) void vq_prep_cnorm(
    const float* __restrict__ cb, float* __restrict__ cnf,
    float* __restrict__ lossOut) {
  int fid = blockIdx.x * 256 + threadIdx.x;            // 0 .. 8191
  int q = fid >> 10, k = fid & 1023;
  const float* row = cb + (size_t)fid * DIM;
  float s = 0.f;
#pragma unroll
  for (int d = 0; d < DIM; ++d) { float v = row[d]; s += v * v; }
  s += 1.0f;   // constant bias: keeps dist strictly > 0, argmin unchanged
  int t = k >> 4, m = k & 15;
  int v = m & 7, l0 = (m >> 3) * 16;
  int base = ((q * 64 + t) * 32 + l0) * 8 + v;
#pragma unroll
  for (int i = 0; i < 16; ++i) cnf[base + i * 8] = s;
  if (fid < QSTAGE) lossOut[fid] = 0.f;
}

// ---------------------------------------------------------------------------
// Main persistent kernel: each wave owns 16 tokens; residual (16x64 f32) lives
// in VGPRs across all 8 stages. Fragments stream through LDS in half-stage
// ping-pong buffers filled by the Tensor Data Mover (wave 0 issues, whole
// block consumes after s_wait_tensorcnt + barrier), overlapping DMA of half
// h+1 with WMMA compute of half h. Distances are seeded with
// ||r||^2 + ||c||^2 + 1 > 0 so u32 ordering == f32 ordering; per tile the 8
// slot keys (low 4 bits = hi*8+v) reduce via a min3 tree, the tile index is
// OR'd into disjoint bits, and one running v_min_u32 carries (dist|index).
// ---------------------------------------------------------------------------
extern __shared__ char smem[];

__global__ __launch_bounds__(256) void vq_main(
    const float* __restrict__ x, const float* __restrict__ cb,
    const unsigned short* __restrict__ afrag, const float* __restrict__ cnf,
    float* __restrict__ xq, float* __restrict__ idxOut,
    float* __restrict__ lossOut) {
  __shared__ float lred[8];

  const int lane = threadIdx.x & 31;
  const int wave = threadIdx.x >> 5;
  const int hi   = lane >> 4;
  const int tok  = (blockIdx.x * 8 + wave) * 16 + (lane & 15);

  // Residual registers: rA = dims [hi*16..+15], rB = dims [32+hi*16..+15]
  float rA[16], rB[16];
  float rn2 = 0.f;                       // this lane's partial ||r||^2
  {
    const f4* p = (const f4*)(x + (size_t)tok * DIM + hi * 16);
#pragma unroll
    for (int i4 = 0; i4 < 4; ++i4) {
      f4 a = p[i4];
      f4 b = p[i4 + 8];
#pragma unroll
      for (int j = 0; j < 4; ++j) {
        rA[i4 * 4 + j] = a[j]; rB[i4 * 4 + j] = b[j];
        rn2 += a[j] * a[j] + b[j] * b[j];
      }
    }
  }

  const float invND = 1.0f / ((float)N_TOK * (float)DIM);

  // per-slot low index bits: hi*8 + v  (loop-invariant, 4 bits)
  unsigned cvi[8];
#pragma unroll
  for (int v = 0; v < 8; ++v) cvi[v] = (unsigned)(hi * 8 + v);

  // Preload half 0 via TDM (wave 0 only; EXEC ignored by tensor ops).
  if (wave == 0) tdm_issue_half(0, smem, afrag, cnf);

  v16bf b0 = {}, b1 = {};
  float rbias = 0.f;
  unsigned best = 0xFFFFFFFFu;

  for (int h = 0; h < 2 * QSTAGE; ++h) {
    const int q  = h >> 1;
    const int hf = h & 1;
    char* buf = smem + (size_t)(h & 1) * BUF_BYTES;

    // Wave 0: issue DMA for the NEXT half into the other buffer (its last
    // readers were separated by the barrier ending half h-1), then wait for
    // the CURRENT half's 2 TDM ops (in-order per wave: <=2 leaves only the
    // just-issued pair outstanding).
    if (wave == 0) {
      if (h + 1 < 2 * QSTAGE) {
        tdm_issue_half(h + 1, smem + (size_t)((h + 1) & 1) * BUF_BYTES,
                       afrag, cnf);
        __builtin_amdgcn_s_wait_tensorcnt(2);
      } else {
        __builtin_amdgcn_s_wait_tensorcnt(0);
      }
    }
    __syncthreads();   // publish half h's fragments to all waves

    if (hf == 0) {
      // new stage: bf16 B fragments of the current residual + seed bias
      v16u b0u, b1u;
#pragma unroll
      for (int i = 0; i < 16; ++i) { b0u[i] = f2bf(rA[i]); b1u[i] = f2bf(rB[i]); }
      b0 = __builtin_bit_cast(v16bf, b0u);
      b1 = __builtin_bit_cast(v16bf, b1u);
      rbias = rn2 + __shfl_xor(rn2, 16, 32);   // +1 bias already in cnf
      best = 0xFFFFFFFFu;
    }

    const char* aP = buf + lane * A_LANE_STRIDE;
    const char* cP = buf + AH_BYTES + lane * C_LANE_STRIDE;
    const unsigned tHalf = (unsigned)hf << 9;   // (hf*32) << 4

    for (int tt = 0; tt < 32; ++tt) {
      A16 a0, a1; C8 cs;
      const char* ab = aP + tt * A_TILE_STRIDE;
      a0.h[0] = *(const u16x8*)(ab);
      a0.h[1] = *(const u16x8*)(ab + 16);
      a1.h[0] = *(const u16x8*)(ab + 32);
      a1.h[1] = *(const u16x8*)(ab + 48);
      const char* cbp = cP + tt * C_TILE_STRIDE;
      cs.h[0] = *(const f4*)(cbp);
      cs.h[1] = *(const f4*)(cbp + 16);
      v8f c = cs.f;
#pragma unroll
      for (int v = 0; v < 8; ++v) c[v] += rbias;   // dist seed > 0

      // dist = ||r||^2 + ||c||^2 + 1 + (-2c).r   (f32 accumulate)
      c = __builtin_amdgcn_wmma_f32_16x16x32_bf16(false, a0.bf, false, b0,
                                                  (short)0, c, false, false);
      c = __builtin_amdgcn_wmma_f32_16x16x32_bf16(false, a1.bf, false, b1,
                                                  (short)0, c, false, false);

      // per-slot keys: (dist & ~0x3FF) | (hi*8 + v)
      CU cu; cu.f = c;
      unsigned k0 = (cu.u[0] & 0xFFFFFC00u) | cvi[0];
      unsigned k1 = (cu.u[1] & 0xFFFFFC00u) | cvi[1];
      unsigned k2 = (cu.u[2] & 0xFFFFFC00u) | cvi[2];
      unsigned k3 = (cu.u[3] & 0xFFFFFC00u) | cvi[3];
      unsigned k4 = (cu.u[4] & 0xFFFFFC00u) | cvi[4];
      unsigned k5 = (cu.u[5] & 0xFFFFFC00u) | cvi[5];
      unsigned k6 = (cu.u[6] & 0xFFFFFC00u) | cvi[6];
      unsigned k7 = (cu.u[7] & 0xFFFFFC00u) | cvi[7];
      unsigned m0 = umin32(umin32(k0, k1), k2);
      unsigned m1 = umin32(umin32(k3, k4), k5);
      unsigned m2 = umin32(umin32(k6, k7), m0);
      unsigned kt = umin32(m1, m2) | (((unsigned)tt << 4) | tHalf);
      best = umin32(best, kt);
    }

    if (hf == 0) {
      __syncthreads();   // all waves done reading buf0-half before TDM reuse
      continue;
    }

    // ---- stage complete: resolve argmin, gather, update, loss ------------
    {
      unsigned ok = __shfl_xor(best, 16, 32);
      best = umin32(best, ok);
    }
    const int bestIdx = (int)(best & 1023u);

    if (hi == 0) idxOut[(size_t)tok * QSTAGE + q] = (float)bestIdx;

    float ls = 0.f;
    {
      const f4* cp = (const f4*)(cb + ((size_t)(q * KCB + bestIdx)) * DIM + hi * 16);
#pragma unroll
      for (int i4 = 0; i4 < 4; ++i4) {
        f4 c0 = cp[i4];
        f4 c1 = cp[i4 + 8];
#pragma unroll
        for (int j = 0; j < 4; ++j) {
          float nA = rA[i4 * 4 + j] - c0[j];
          float nB = rB[i4 * 4 + j] - c1[j];
          rA[i4 * 4 + j] = nA;
          rB[i4 * 4 + j] = nB;
          ls += nA * nA + nB * nB;   // (quant - residual)^2 == new_residual^2
        }
      }
    }
    rn2 = ls;   // reuse as next stage's ||r||^2 partial

    // loss reduction: wave shfl tree -> LDS across 8 waves -> 1 atomic/block
#pragma unroll
    for (int s = 16; s > 0; s >>= 1) ls += __shfl_xor(ls, s, 32);
    if (lane == 0) lred[wave] = ls;
    __syncthreads();
    if (threadIdx.x == 0) {
      float bs = 0.f;
#pragma unroll
      for (int w = 0; w < 8; ++w) bs += lred[w];
      atomicAdd(lossOut + q, bs * invND);
    }
    __syncthreads();   // also protects this half's buffer before TDM reuse
  }

  // xq = x - final_residual (telescoping sum of per-stage quants)
  {
    const f4* p = (const f4*)(x  + (size_t)tok * DIM + hi * 16);
    f4*       o = (f4*)      (xq + (size_t)tok * DIM + hi * 16);
#pragma unroll
    for (int i4 = 0; i4 < 4; ++i4) {
      f4 xa = p[i4], xb = p[i4 + 8];
      f4 ra, rb;
#pragma unroll
      for (int j = 0; j < 4; ++j) { ra[j] = rA[i4 * 4 + j]; rb[j] = rB[i4 * 4 + j]; }
      o[i4]     = xa - ra;
      o[i4 + 8] = xb - rb;
    }
  }
}

// ---------------------------------------------------------------------------
// Launch
// ---------------------------------------------------------------------------
extern "C" void kernel_launch(void* const* d_in, const int* in_sizes, int n_in,
                              void* d_out, int out_size, void* d_ws,
                              size_t ws_size, hipStream_t stream) {
  const float* x  = (const float*)d_in[0];   // (N, D) f32
  const float* cb = (const float*)d_in[1];   // (Q, K, D) f32

  float* out     = (float*)d_out;
  float* xq      = out;                                   // N*D
  float* idxOut  = out + (size_t)N_TOK * DIM;             // N*Q (as float)
  float* lossOut = idxOut + (size_t)N_TOK * QSTAGE;       // Q

  unsigned short* afrag = (unsigned short*)d_ws;                    // 1 MB
  float*          cnf   = (float*)((char*)d_ws + (1u << 20));       // 512 KB

  vq_prep_afrag<<<2048, 256, 0, stream>>>(cb, afrag);
  vq_prep_cnorm<<<32, 256, 0, stream>>>(cb, cnf, lossOut);
  vq_main<<<N_TOK / 128, 256, SMEM_BYTES, stream>>>(x, cb, afrag, cnf,
                                                    xq, idxOut, lossOut);
}
